// HexTransformerLayer_27419071218358
// MI455X (gfx1250) — matrix operand order repaired and checked by reading, not compiled, not run on hardware
//
#include <hip/hip_runtime.h>
#include <hip/hip_bf16.h>
#include <math.h>
#include <stdint.h>

#define BB 4
#define NN 2048
#define DD 1024
#define HH 16
#define DHH 64
#define DFF_ 4096
#define BN 8192

typedef __bf16 v16bf __attribute__((ext_vector_type(16)));
typedef float  v8f   __attribute__((ext_vector_type(8)));

union FragBF { v16bf v; uint4 q[2]; };
union FragF  { v8f v; float f[8]; };

__device__ __forceinline__ unsigned short f2bf(float f) {
  unsigned u = __float_as_uint(f);
  u += 0x7FFFu + ((u >> 16) & 1u);           // round-to-nearest-even
  return (unsigned short)(u >> 16);
}

// CDNA5 async global->LDS copy (ASYNCcnt-tracked, no VGPR data round-trip).
// lds_off: byte offset of destination within the workgroup LDS allocation
// (low 32 bits of the generic pointer, per ISA flat-LDS addressing).
__device__ __forceinline__ void async_copy_b128(unsigned lds_off, const void* g) {
  asm volatile("global_load_async_to_lds_b128 %0, %1, off"
               :: "v"(lds_off), "v"((unsigned long long)g) : "memory");
}
__device__ __forceinline__ void async_wait0() {
  asm volatile("s_wait_asynccnt 0x0" ::: "memory");
}

// ---------------------------------------------------------------------------
// Weight fp32 (K x N) -> bf16 transposed (N x K), LDS-tiled transpose
// ---------------------------------------------------------------------------
__global__ __launch_bounds__(256) void wtrans_kernel(const float* __restrict__ W,
                                                     unsigned short* __restrict__ Wt,
                                                     int K, int N) {
  __shared__ float tile[32][33];
  int n0 = blockIdx.x * 32, k0 = blockIdx.y * 32;
  int tx = threadIdx.x, ty = threadIdx.y;
#pragma unroll
  for (int r = 0; r < 4; ++r)
    tile[ty + r * 8][tx] = W[(size_t)(k0 + ty + r * 8) * N + (n0 + tx)];
  __syncthreads();
#pragma unroll
  for (int r = 0; r < 4; ++r)
    Wt[(size_t)(n0 + ty + r * 8) * K + (k0 + tx)] = f2bf(tile[tx][ty + r * 8]);
}

// ---------------------------------------------------------------------------
// LayerNorm over D=1024, fp32 in -> bf16 out (one block per row, 8 waves)
// ---------------------------------------------------------------------------
__global__ __launch_bounds__(256) void ln_kernel(const float* __restrict__ x,
                                                 const float* __restrict__ g,
                                                 const float* __restrict__ b,
                                                 unsigned short* __restrict__ out) {
  int row = blockIdx.x, t = threadIdx.x;
  const float* xr = x + (size_t)row * DD;
  float4 v = *(const float4*)(xr + t * 4);
  float s = v.x + v.y + v.z + v.w;
  float ss = v.x * v.x + v.y * v.y + v.z * v.z + v.w * v.w;
#pragma unroll
  for (int off = 16; off > 0; off >>= 1) {
    s  += __shfl_xor(s, off, 32);
    ss += __shfl_xor(ss, off, 32);
  }
  __shared__ float sh_s[8], sh_q[8];
  int wid = t >> 5;
  if ((t & 31) == 0) { sh_s[wid] = s; sh_q[wid] = ss; }
  __syncthreads();
  s = 0.f; ss = 0.f;
#pragma unroll
  for (int i = 0; i < 8; ++i) { s += sh_s[i]; ss += sh_q[i]; }
  float mean = s * (1.f / DD);
  float var  = ss * (1.f / DD) - mean * mean;
  float rstd = rsqrtf(var + 1e-5f);
  float4 gv = *(const float4*)(g + t * 4);
  float4 bv = *(const float4*)(b + t * 4);
  unsigned o0 = f2bf((v.x - mean) * rstd * gv.x + bv.x);
  unsigned o1 = f2bf((v.y - mean) * rstd * gv.y + bv.y);
  unsigned o2 = f2bf((v.z - mean) * rstd * gv.z + bv.z);
  unsigned o3 = f2bf((v.w - mean) * rstd * gv.w + bv.w);
  uint2 pk;
  pk.x = o0 | (o1 << 16);
  pk.y = o2 | (o3 << 16);
  *(uint2*)(out + (size_t)row * DD + t * 4) = pk;
}

// ---------------------------------------------------------------------------
// bf16 WMMA GEMM: C(MxN) = A(MxK) * Wt(NxK)^T + bias, fused epilogues.
// 128x128 block tile, 8 waves (each 64x32), K-step 32.
// Double-buffered LDS fed by GLOBAL_LOAD_ASYNC_TO_LDS_B128 (ASYNCcnt).
// ---------------------------------------------------------------------------
enum { EPI_QKV = 0, EPI_PROJ = 1, EPI_FF1 = 2, EPI_FF2 = 3 };

template <int EPI>
__global__ __launch_bounds__(256, 1) void gemm_bf16_kernel(
    const unsigned short* __restrict__ A,    // M x K bf16
    const unsigned short* __restrict__ Wt,   // N x K bf16 (pre-transposed)
    const float* __restrict__ bias,          // N
    const float* __restrict__ resid,         // M x N fp32 (PROJ/FF2)
    float* __restrict__ outF,                // M x N fp32 (PROJ/FF2)
    unsigned short* __restrict__ outQ,       // QKV outputs
    unsigned short* __restrict__ outK,
    unsigned short* __restrict__ outV,
    unsigned short* __restrict__ outBf,      // FF1 bf16 output
    int M, int N, int K) {
  __shared__ unsigned short As[2][128][40];
  __shared__ unsigned short Bs[2][128][40];

  int t = threadIdx.x;
  int lane = t & 31, wid = t >> 5;
  int half = lane >> 4, l16 = lane & 15;
  int wm = wid >> 2, wn = wid & 3;
  int m0 = blockIdx.y * 128, n0 = blockIdx.x * 128;

  int lrow = t >> 1, lseg = (t & 1) * 16;
  const unsigned short* aP = A  + (size_t)(m0 + lrow) * K + lseg;
  const unsigned short* bP = Wt + (size_t)(n0 + lrow) * K + lseg;

  auto stage = [&](int buf, int kt) {
    const unsigned short* ap = aP + (size_t)kt * 32;
    const unsigned short* bp = bP + (size_t)kt * 32;
    unsigned da = (unsigned)(uintptr_t)&As[buf][lrow][lseg];
    unsigned db = (unsigned)(uintptr_t)&Bs[buf][lrow][lseg];
    async_copy_b128(da,      ap);
    async_copy_b128(da + 16, ap + 8);
    async_copy_b128(db,      bp);
    async_copy_b128(db + 16, bp + 8);
  };

  stage(0, 0);
  async_wait0();
  __syncthreads();

  FragF acc[4][2];
#pragma unroll
  for (int mi = 0; mi < 4; ++mi)
#pragma unroll
    for (int ni = 0; ni < 2; ++ni)
#pragma unroll
      for (int j = 0; j < 8; ++j) acc[mi][ni].f[j] = 0.f;

  int nk = K >> 5;
  for (int kt = 0; kt < nk; ++kt) {
    int cur = kt & 1;
    bool next = (kt + 1) < nk;
    if (next) stage(cur ^ 1, kt + 1);   // async copy overlaps WMMA below

    FragBF af[4], bfr[2];
#pragma unroll
    for (int mi = 0; mi < 4; ++mi) {
      const unsigned short* p = &As[cur][wm * 64 + mi * 16 + l16][half * 8];
      af[mi].q[0] = *(const uint4*)p;
      af[mi].q[1] = *(const uint4*)(p + 16);
    }
#pragma unroll
    for (int ni = 0; ni < 2; ++ni) {
      const unsigned short* p = &Bs[cur][wn * 32 + ni * 16 + l16][half * 8];
      bfr[ni].q[0] = *(const uint4*)p;
      bfr[ni].q[1] = *(const uint4*)(p + 16);
    }
#pragma unroll
    for (int mi = 0; mi < 4; ++mi)
#pragma unroll
      for (int ni = 0; ni < 2; ++ni)
        acc[mi][ni].v = __builtin_amdgcn_wmma_f32_16x16x32_bf16(
            false, af[mi].v, false, bfr[ni].v, (short)0, acc[mi][ni].v, false, false);

    if (next) async_wait0();
    __syncthreads();
  }

  // epilogue: C layout lane(i): row = i + half*8, col = l16
#pragma unroll
  for (int mi = 0; mi < 4; ++mi)
#pragma unroll
    for (int ni = 0; ni < 2; ++ni)
#pragma unroll
      for (int i = 0; i < 8; ++i) {
        int R = m0 + wm * 64 + mi * 16 + i + half * 8;
        int C = n0 + wn * 32 + ni * 16 + l16;
        float v = acc[mi][ni].f[i] + bias[C];
        if (EPI == EPI_QKV) {
          int which = C >> 10, h = (C >> 6) & 15, dh = C & 63;
          int b_ = R >> 11, n_ = R & 2047;
          if (which == 0)       // Q, pre-scaled by 1/sqrt(DH)
            outQ[(((size_t)(b_ * HH + h)) * NN + n_) * DHH + dh] = f2bf(v * 0.125f);
          else if (which == 1)  // K: (B,H,N,DH)
            outK[(((size_t)(b_ * HH + h)) * NN + n_) * DHH + dh] = f2bf(v);
          else                  // V: transposed (B,H,DH,N) for P@V B-fragments
            outV[(((size_t)(b_ * HH + h)) * DHH + dh) * NN + n_] = f2bf(v);
        } else if (EPI == EPI_PROJ) {
          outF[(size_t)R * N + C] = v + resid[(size_t)R * N + C];
        } else if (EPI == EPI_FF1) {
          float gl = 0.5f * v * (1.f + erff(v * 0.70710678118f));
          outBf[(size_t)R * N + C] = f2bf(gl);
        } else {  // EPI_FF2
          outF[(size_t)R * N + C] = v + resid[(size_t)R * N + C];
        }
      }
}

// ---------------------------------------------------------------------------
// Flash attention: block = 128 query rows x all 2048 keys (tiles of 128).
// 8 waves; wave handles 16 query rows. WMMA for QK^T and P@V.
// K/V tiles staged with async global->LDS copies.
// ---------------------------------------------------------------------------
__global__ __launch_bounds__(256, 1) void attn_kernel(
    const unsigned short* __restrict__ Q,    // (B*H, N, DH) bf16, pre-scaled
    const unsigned short* __restrict__ Kb,   // (B*H, N, DH) bf16
    const unsigned short* __restrict__ Vt,   // (B*H, DH, N) bf16
    const float* __restrict__ bias,          // (H, N, N) fp32
    unsigned short* __restrict__ O) {        // (B*N, D) bf16
  __shared__ unsigned short ks[128][72];
  __shared__ unsigned short vs[64][136];
  __shared__ unsigned short ps[8][16][136];

  int t = threadIdx.x;
  int lane = t & 31, wid = t >> 5;
  int half = lane >> 4, l16 = lane & 15;
  int bh = blockIdx.y;
  int b_ = bh >> 4, h = bh & 15;
  int q0 = blockIdx.x * 128;
  int qrow = q0 + wid * 16 + l16;

  FragBF qa[2];
#pragma unroll
  for (int s = 0; s < 2; ++s) {
    const unsigned short* p = Q + ((size_t)bh * NN + qrow) * DHH + s * 32 + half * 8;
    qa[s].q[0] = *(const uint4*)p;
    qa[s].q[1] = *(const uint4*)(p + 16);
  }

  FragF oacc[4];
  float m_i[8], l_i[8];
#pragma unroll
  for (int dt = 0; dt < 4; ++dt)
#pragma unroll
    for (int j = 0; j < 8; ++j) oacc[dt].f[j] = 0.f;
#pragma unroll
  for (int i = 0; i < 8; ++i) { m_i[i] = -3.0e38f; l_i[i] = 0.f; }

  const float* bp = bias + ((size_t)h * NN + q0 + wid * 16 + half * 8) * NN + l16;

  // per-thread staging addresses (fixed across key tiles)
  unsigned kd = (unsigned)(uintptr_t)&ks[t >> 1][(t & 1) * 32];
  unsigned vd = (unsigned)(uintptr_t)&vs[t >> 2][(t & 3) * 32];

  for (int nt0 = 0; nt0 < NN; nt0 += 128) {
    __syncthreads();
    {  // stage K tile: 128 keys x 64 dh (async)
      const unsigned short* p = Kb + ((size_t)bh * NN + nt0 + (t >> 1)) * DHH + (t & 1) * 32;
      async_copy_b128(kd,      p);
      async_copy_b128(kd + 16, p + 8);
      async_copy_b128(kd + 32, p + 16);
      async_copy_b128(kd + 48, p + 24);
    }
    {  // stage V tile (dh-major): 64 dh x 128 keys (async)
      const unsigned short* p = Vt + ((size_t)bh * DHH + (t >> 2)) * NN + nt0 + (t & 3) * 32;
      async_copy_b128(vd,      p);
      async_copy_b128(vd + 16, p + 8);
      async_copy_b128(vd + 32, p + 16);
      async_copy_b128(vd + 48, p + 24);
    }
    async_wait0();
    __syncthreads();

    // S = q @ k^T  (16 x 128 per wave, 8 key sub-tiles)
    FragF sacc[8];
#pragma unroll
    for (int nt = 0; nt < 8; ++nt) {
#pragma unroll
      for (int j = 0; j < 8; ++j) sacc[nt].f[j] = 0.f;
#pragma unroll
      for (int s = 0; s < 2; ++s) {
        FragBF kf;
        const unsigned short* p = &ks[nt * 16 + l16][s * 32 + half * 8];
        kf.q[0] = *(const uint4*)p;
        kf.q[1] = *(const uint4*)(p + 16);
        sacc[nt].v = __builtin_amdgcn_wmma_f32_16x16x32_bf16(
            false, qa[s].v, false, kf.v, (short)0, sacc[nt].v, false, false);
      }
    }

    // + structural bias, per-row running max
    float mt[8];
#pragma unroll
    for (int i = 0; i < 8; ++i) mt[i] = -3.0e38f;
#pragma unroll
    for (int nt = 0; nt < 8; ++nt)
#pragma unroll
      for (int i = 0; i < 8; ++i) {
        float sv = sacc[nt].f[i] + bp[(size_t)i * NN + nt0 + nt * 16];
        sacc[nt].f[i] = sv;
        mt[i] = fmaxf(mt[i], sv);
      }
#pragma unroll
    for (int i = 0; i < 8; ++i)
#pragma unroll
      for (int off = 1; off < 16; off <<= 1)
        mt[i] = fmaxf(mt[i], __shfl_xor(mt[i], off, 32));

    float sc[8];
#pragma unroll
    for (int i = 0; i < 8; ++i) {
      float mn = fmaxf(m_i[i], mt[i]);
      sc[i] = __expf(m_i[i] - mn);
      m_i[i] = mn;
    }
    float rs[8];
#pragma unroll
    for (int i = 0; i < 8; ++i) rs[i] = 0.f;
#pragma unroll
    for (int nt = 0; nt < 8; ++nt)
#pragma unroll
      for (int i = 0; i < 8; ++i) {
        float pe = __expf(sacc[nt].f[i] - m_i[i]);
        sacc[nt].f[i] = pe;
        rs[i] += pe;
      }
#pragma unroll
    for (int i = 0; i < 8; ++i) {
#pragma unroll
      for (int off = 1; off < 16; off <<= 1)
        rs[i] += __shfl_xor(rs[i], off, 32);
      l_i[i] = l_i[i] * sc[i] + rs[i];
    }
#pragma unroll
    for (int dt = 0; dt < 4; ++dt)
#pragma unroll
      for (int i = 0; i < 8; ++i) oacc[dt].f[i] *= sc[i];

    // P (bf16) -> per-wave LDS, then re-read in A-fragment layout
#pragma unroll
    for (int nt = 0; nt < 8; ++nt)
#pragma unroll
      for (int i = 0; i < 8; ++i)
        ps[wid][i + half * 8][nt * 16 + l16] = f2bf(sacc[nt].f[i]);
    asm volatile("s_wait_dscnt 0x0" ::: "memory");  // wave-private region: DS wait suffices

    // O += P @ V
#pragma unroll
    for (int s = 0; s < 4; ++s) {
      FragBF pf;
      const unsigned short* pp = &ps[wid][l16][s * 32 + half * 8];
      pf.q[0] = *(const uint4*)pp;
      pf.q[1] = *(const uint4*)(pp + 16);
#pragma unroll
      for (int dt = 0; dt < 4; ++dt) {
        FragBF vf;
        const unsigned short* vp = &vs[dt * 16 + l16][s * 32 + half * 8];
        vf.q[0] = *(const uint4*)vp;
        vf.q[1] = *(const uint4*)(vp + 16);
        oacc[dt].v = __builtin_amdgcn_wmma_f32_16x16x32_bf16(
            false, pf.v, false, vf.v, (short)0, oacc[dt].v, false, false);
      }
    }
  }

  // normalize, store as (B, N, H*DH) bf16 -> GEMM-ready activation
#pragma unroll
  for (int i = 0; i < 8; ++i) l_i[i] = 1.f / l_i[i];
#pragma unroll
  for (int dt = 0; dt < 4; ++dt)
#pragma unroll
    for (int i = 0; i < 8; ++i) {
      int n_ = q0 + wid * 16 + i + half * 8;
      int c  = h * DHH + dt * 16 + l16;
      O[((size_t)b_ * NN + n_) * DD + c] = f2bf(oacc[dt].f[i] * l_i[i]);
    }
}

// ---------------------------------------------------------------------------
extern "C" void kernel_launch(void* const* d_in, const int* in_sizes, int n_in,
                              void* d_out, int out_size, void* d_ws, size_t ws_size,
                              hipStream_t stream) {
  const float* x    = (const float*)d_in[0];
  const float* sb   = (const float*)d_in[1];
  const float* ln1g = (const float*)d_in[2];
  const float* ln1b = (const float*)d_in[3];
  const float* ln2g = (const float*)d_in[4];
  const float* ln2b = (const float*)d_in[5];
  const float* Wqkv = (const float*)d_in[6];
  const float* bqkv = (const float*)d_in[7];
  const float* Wout = (const float*)d_in[8];
  const float* bout = (const float*)d_in[9];
  const float* Wff1 = (const float*)d_in[10];
  const float* bff1 = (const float*)d_in[11];
  const float* Wff2 = (const float*)d_in[12];
  const float* bff2 = (const float*)d_in[13];
  (void)in_sizes; (void)n_in; (void)out_size; (void)ws_size;

  char* w = (char*)d_ws;
  size_t off = 0;
  auto carve = [&](size_t bytes) -> char* {
    char* p = w + off;
    off += (bytes + 255) & ~(size_t)255;
    return p;
  };
  unsigned short* WqkvT = (unsigned short*)carve((size_t)3 * DD * DD * 2);
  unsigned short* WoutT = (unsigned short*)carve((size_t)DD * DD * 2);
  unsigned short* Wff1T = (unsigned short*)carve((size_t)DD * DFF_ * 2);
  unsigned short* Wff2T = (unsigned short*)carve((size_t)DFF_ * DD * 2);
  unsigned short* A1    = (unsigned short*)carve((size_t)BN * DD * 2);
  unsigned short* Qb    = (unsigned short*)carve((size_t)BN * DD * 2);
  unsigned short* Kbuf  = (unsigned short*)carve((size_t)BN * DD * 2);
  unsigned short* VtB   = (unsigned short*)carve((size_t)BN * DD * 2);
  unsigned short* AttnA = (unsigned short*)carve((size_t)BN * DD * 2);
  float*          X1    = (float*)carve((size_t)BN * DD * 4);
  unsigned short* A2    = (unsigned short*)carve((size_t)BN * DD * 2);
  unsigned short* Hb    = (unsigned short*)carve((size_t)BN * DFF_ * 2);

  dim3 tb(32, 8);
  wtrans_kernel<<<dim3(3 * DD / 32, DD / 32), tb, 0, stream>>>(Wqkv, WqkvT, DD, 3 * DD);
  wtrans_kernel<<<dim3(DD / 32, DD / 32), tb, 0, stream>>>(Wout, WoutT, DD, DD);
  wtrans_kernel<<<dim3(DFF_ / 32, DD / 32), tb, 0, stream>>>(Wff1, Wff1T, DD, DFF_);
  wtrans_kernel<<<dim3(DD / 32, DFF_ / 32), tb, 0, stream>>>(Wff2, Wff2T, DFF_, DD);

  ln_kernel<<<BN, 256, 0, stream>>>(x, ln1g, ln1b, A1);

  gemm_bf16_kernel<EPI_QKV><<<dim3(3 * DD / 128, BN / 128), 256, 0, stream>>>(
      A1, WqkvT, bqkv, nullptr, nullptr, Qb, Kbuf, VtB, nullptr, BN, 3 * DD, DD);

  attn_kernel<<<dim3(NN / 128, BB * HH), 256, 0, stream>>>(Qb, Kbuf, VtB, sb, AttnA);

  gemm_bf16_kernel<EPI_PROJ><<<dim3(DD / 128, BN / 128), 256, 0, stream>>>(
      AttnA, WoutT, bout, x, X1, nullptr, nullptr, nullptr, nullptr, BN, DD, DD);

  ln_kernel<<<BN, 256, 0, stream>>>(X1, ln2g, ln2b, A2);

  gemm_bf16_kernel<EPI_FF1><<<dim3(DFF_ / 128, BN / 128), 256, 0, stream>>>(
      A2, Wff1T, bff1, nullptr, nullptr, nullptr, nullptr, nullptr, Hb, BN, DFF_, DD);

  gemm_bf16_kernel<EPI_FF2><<<dim3(DD / 128, BN / 128), 256, 0, stream>>>(
      Hb, Wff2T, bff2, X1, (float*)d_out, nullptr, nullptr, nullptr, nullptr, BN, DD, DFF_);
}
